// CausalSelfAttention_37271726195022
// MI455X (gfx1250) — matrix operand order repaired and checked by reading, not compiled, not run on hardware
//
#include <hip/hip_runtime.h>

typedef __attribute__((ext_vector_type(16))) __bf16 v16bf;
typedef __attribute__((ext_vector_type(2)))  __bf16 v2bf;
typedef __attribute__((ext_vector_type(8)))  float  v8f;

union FragAB { v16bf v; unsigned int u[8]; };

// native RNE fp32 -> bf16 (lowers to v_cvt_*bf16* on gfx1250)
__device__ __forceinline__ unsigned short f2bf(float f) {
    __bf16 h = (__bf16)f;
    return __builtin_bit_cast(unsigned short, h);
}

// packed pair: two fp32 -> one dword of two bf16 (v_cvt_pk_bf16_f32 shape)
__device__ __forceinline__ unsigned int pk2bf(float lo, float hi) {
    v2bf p;
    p.x = (__bf16)lo;
    p.y = (__bf16)hi;
    return __builtin_bit_cast(unsigned int, p);
}

__device__ __forceinline__ v8f wmma_bf16(const FragAB& a, const FragAB& b, v8f c) {
    return __builtin_amdgcn_wmma_f32_16x16x32_bf16(false, a.v, false, b.v,
                                                   (short)0, c, false, false);
}

// ---------------------------------------------------------------------------
// Kernel 1: qkv = x @ W_attn + b_attn   (M=8192, N=3072, K=1024)
// Writes bf16: q,k as [b,h,t,hd] (q pre-scaled by 1/sqrt(64)), v as [b,h,hd,t]
// ---------------------------------------------------------------------------
__global__ __launch_bounds__(128) void qkv_gemm_kernel(
    const float* __restrict__ x, const float* __restrict__ W,
    const float* __restrict__ bias,
    unsigned short* __restrict__ qd, unsigned short* __restrict__ kd,
    unsigned short* __restrict__ vd)
{
    __shared__ unsigned short ldsB[64][34];     // B tile transposed [N][K], padded
    const int tid  = threadIdx.x;
    const int lane = tid & 31;
    const int w    = tid >> 5;
    const int bn   = blockIdx.x % 48;
    const int bm   = blockIdx.x / 48;
    const int n0   = bn * 64;
    const int mw   = bm * 64 + w * 16;
    const int lm   = lane & 15;
    const int hi   = lane >> 4;

    v8f zero = {};
    v8f acc[4] = {zero, zero, zero, zero};

    for (int k0 = 0; k0 < 1024; k0 += 32) {
        __syncthreads();
        #pragma unroll
        for (int i = 0; i < 16; ++i) {
            int e  = tid + 128 * i;
            int kk = e >> 6;                     // 0..31
            int nn = e & 63;                     // 0..63
            ldsB[nn][kk] = f2bf(W[(size_t)(k0 + kk) * 3072 + n0 + nn]);
        }
        __syncthreads();
        // A fragment: direct from global fp32 (b64 pair loads), packed convert
        FragAB a;
        const float* xr = x + (size_t)(mw + lm) * 1024 + k0;
        #pragma unroll
        for (int j = 0; j < 8; ++j) {
            int kp = hi * 16 + 2 * j;
            float2 f = *(const float2*)(xr + kp);
            a.u[j] = pk2bf(f.x, f.y);
        }
        #pragma unroll
        for (int t = 0; t < 4; ++t) {
            FragAB bf;
            int nn = t * 16 + lm;
            int ko = hi * 16;
            #pragma unroll
            for (int j = 0; j < 8; ++j)
                bf.u[j] = *(const unsigned int*)&ldsB[nn][ko + 2 * j];
            acc[t] = wmma_bf16(a, bf, acc[t]);
        }
    }
    // Epilogue: bias, split into q/k/v, reshape to heads
    #pragma unroll
    for (int t = 0; t < 4; ++t) {
        int n = n0 + t * 16 + lm;
        float bv = bias[n];
        int which = n >> 10;                     // 0:q 1:k 2:v
        int c  = n & 1023;
        int h  = c >> 6;
        int hd = c & 63;
        #pragma unroll
        for (int r = 0; r < 8; ++r) {
            int m  = mw + r + 8 * hi;
            int bb = m >> 11;
            int tq = m & 2047;
            float val = acc[t][r] + bv;
            size_t bh = (size_t)(bb * 16 + h);
            if (which == 0)      qd[(bh * 2048 + tq) * 64 + hd] = f2bf(val * 0.125f);
            else if (which == 1) kd[(bh * 2048 + tq) * 64 + hd] = f2bf(val);
            else                 vd[(bh * 64 + hd) * 2048 + tq] = f2bf(val);
        }
    }
}

// ---------------------------------------------------------------------------
// Kernel 2: causal flash attention, one wave per (b, h, 16-query block)
// ---------------------------------------------------------------------------
__global__ __launch_bounds__(256) void attn_kernel(
    const unsigned short* __restrict__ qd, const unsigned short* __restrict__ kd,
    const unsigned short* __restrict__ vd, unsigned short* __restrict__ yd)
{
    __shared__ unsigned short ldsP[8][16][34];  // per-wave P staging (16x32 bf16)
    const int tid  = threadIdx.x;
    const int lane = tid & 31;
    const int w    = tid >> 5;
    const int gw   = blockIdx.x * 8 + w;        // 0..8191
    const int qb   = gw & 127;
    const int h    = (gw >> 7) & 15;
    const int b    = gw >> 11;
    const int q0   = qb * 16;
    const size_t bh = (size_t)(b * 16 + h);
    const unsigned short* Qp = qd + bh * 2048 * 64;
    const unsigned short* Kp = kd + bh * 2048 * 64;
    const unsigned short* Vp = vd + bh * 2048 * 64;   // [hd][t]
    unsigned short*       Yp = yd + bh * 2048 * 64;
    const int lm = lane & 15, hi = lane >> 4;

    // Q A-fragments for hd 0-31 and 32-63 (scale already folded in)
    FragAB aq[2];
    {
        const unsigned short* qr = Qp + (size_t)(q0 + lm) * 64;
        #pragma unroll
        for (int s = 0; s < 2; ++s)
            #pragma unroll
            for (int j = 0; j < 8; ++j) {
                int kp = 32 * s + hi * 16 + 2 * j;
                aq[s].u[j] = *(const unsigned int*)(qr + kp);
            }
    }
    v8f zero = {};
    v8f o[4] = {zero, zero, zero, zero};
    float mrow[8], lrow[8];
    #pragma unroll
    for (int r = 0; r < 8; ++r) { mrow[r] = -__builtin_inff(); lrow[r] = 0.f; }

    const int nblocks = (q0 + 16 + 31) >> 5;    // key blocks of 32 up to diagonal
    for (int kb = 0; kb < nblocks; ++kb) {
        const int kbase = kb * 32;
        v8f s0 = zero, s1 = zero;
        #pragma unroll
        for (int c = 0; c < 2; ++c) {           // contraction halves over hd
            FragAB bk0, bk1;
            const unsigned short* kr0 = Kp + (size_t)(kbase + lm) * 64      + 32 * c + hi * 16;
            const unsigned short* kr1 = Kp + (size_t)(kbase + 16 + lm) * 64 + 32 * c + hi * 16;
            #pragma unroll
            for (int j = 0; j < 8; ++j) {
                bk0.u[j] = *(const unsigned int*)(kr0 + 2 * j);
                bk1.u[j] = *(const unsigned int*)(kr1 + 2 * j);
            }
            s0 = wmma_bf16(aq[c], bk0, s0);
            s1 = wmma_bf16(aq[c], bk1, s1);
        }
        // online softmax (rows live in 16-lane halves: M = r + 8*hi)
        float p0[8], p1[8];
        #pragma unroll
        for (int r = 0; r < 8; ++r) {
            const int qrow = q0 + r + 8 * hi;
            float v0 = s0[r]; if (kbase + lm > qrow)      v0 = -1e30f;
            float v1 = s1[r]; if (kbase + 16 + lm > qrow) v1 = -1e30f;
            float mx = fmaxf(v0, v1);
            #pragma unroll
            for (int msk = 1; msk < 16; msk <<= 1)
                mx = fmaxf(mx, __shfl_xor(mx, msk, 32));
            float mnew  = fmaxf(mrow[r], mx);
            float alpha = __expf(mrow[r] - mnew);
            mrow[r] = mnew;
            float e0 = __expf(v0 - mnew);
            float e1 = __expf(v1 - mnew);
            float rs = e0 + e1;
            #pragma unroll
            for (int msk = 1; msk < 16; msk <<= 1)
                rs += __shfl_xor(rs, msk, 32);
            lrow[r] = lrow[r] * alpha + rs;
            p0[r] = e0; p1[r] = e1;
            #pragma unroll
            for (int t = 0; t < 4; ++t) o[t][r] *= alpha;
        }
        // P: C-layout regs -> LDS bf16 (16-bit stores, dword reads) -> A-layout frag
        #pragma unroll
        for (int r = 0; r < 8; ++r) {
            ldsP[w][r + 8 * hi][lm]      = f2bf(p0[r]);
            ldsP[w][r + 8 * hi][16 + lm] = f2bf(p1[r]);
        }
        asm volatile("s_wait_dscnt 0x0" ::: "memory");
        FragAB pa;
        #pragma unroll
        for (int j = 0; j < 8; ++j) {
            int kp = hi * 16 + 2 * j;
            pa.u[j] = *(const unsigned int*)&ldsP[w][lm][kp];
        }
        // O += P @ V  (V transposed [hd][t]: key pairs contiguous)
        #pragma unroll
        for (int t = 0; t < 4; ++t) {
            FragAB bv;
            const unsigned short* vr = Vp + (size_t)(t * 16 + lm) * 2048 + kbase + hi * 16;
            #pragma unroll
            for (int j = 0; j < 8; ++j)
                bv.u[j] = *(const unsigned int*)(vr + 2 * j);
            o[t] = wmma_bf16(pa, bv, o[t]);
        }
    }
    // normalize and store y [b,h,t,hd] bf16
    #pragma unroll
    for (int t = 0; t < 4; ++t) {
        int hd = t * 16 + lm;
        #pragma unroll
        for (int r = 0; r < 8; ++r) {
            int tq = q0 + r + 8 * hi;
            Yp[(size_t)tq * 64 + hd] = f2bf(o[t][r] / lrow[r]);
        }
    }
}

// ---------------------------------------------------------------------------
// Kernel 3: out = y @ W_proj + b_proj   (M=8192, N=1024, K=1024), fp32 out
// ---------------------------------------------------------------------------
__global__ __launch_bounds__(128) void proj_gemm_kernel(
    const unsigned short* __restrict__ y, const float* __restrict__ W,
    const float* __restrict__ bias, float* __restrict__ out)
{
    __shared__ unsigned short ldsB[64][34];
    const int tid  = threadIdx.x;
    const int lane = tid & 31;
    const int w    = tid >> 5;
    const int bn   = blockIdx.x % 16;
    const int bm   = blockIdx.x / 16;
    const int n0   = bn * 64;
    const int mw   = bm * 64 + w * 16;
    const int lm   = lane & 15, hi = lane >> 4;

    v8f zero = {};
    v8f acc[4] = {zero, zero, zero, zero};

    const int m  = mw + lm;
    const int bb = m >> 11;
    const int tq = m & 2047;

    for (int k0 = 0; k0 < 1024; k0 += 32) {
        __syncthreads();
        #pragma unroll
        for (int i = 0; i < 16; ++i) {
            int e  = tid + 128 * i;
            int kk = e >> 6;
            int nn = e & 63;
            ldsB[nn][kk] = f2bf(W[(size_t)(k0 + kk) * 1024 + n0 + nn]);
        }
        __syncthreads();
        // A fragment from bf16 y [b,h,t,hd]; c = h*64+hd, pairs stay within head
        FragAB a;
        #pragma unroll
        for (int j = 0; j < 8; ++j) {
            int kp = hi * 16 + 2 * j;
            int kg = k0 + kp;
            int hh = kg >> 6, hd = kg & 63;
            a.u[j] = *(const unsigned int*)
                     &y[(((size_t)(bb * 16 + hh)) * 2048 + tq) * 64 + hd];
        }
        #pragma unroll
        for (int t = 0; t < 4; ++t) {
            FragAB bf;
            int nn = t * 16 + lm, ko = hi * 16;
            #pragma unroll
            for (int j = 0; j < 8; ++j)
                bf.u[j] = *(const unsigned int*)&ldsB[nn][ko + 2 * j];
            acc[t] = wmma_bf16(a, bf, acc[t]);
        }
    }
    #pragma unroll
    for (int t = 0; t < 4; ++t) {
        int n = n0 + t * 16 + lm;
        float bv = bias[n];
        #pragma unroll
        for (int r = 0; r < 8; ++r) {
            int mm = mw + r + 8 * hi;
            out[(size_t)mm * 1024 + n] = acc[t][r] + bv;
        }
    }
}

// ---------------------------------------------------------------------------
extern "C" void kernel_launch(void* const* d_in, const int* in_sizes, int n_in,
                              void* d_out, int out_size, void* d_ws, size_t ws_size,
                              hipStream_t stream)
{
    const float* x      = (const float*)d_in[0];
    const float* W_attn = (const float*)d_in[1];
    const float* b_attn = (const float*)d_in[2];
    const float* W_proj = (const float*)d_in[3];
    const float* b_proj = (const float*)d_in[4];
    float* out = (float*)d_out;

    const size_t elems = (size_t)8192 * 1024;   // B*T*C
    unsigned short* qd = (unsigned short*)d_ws; // 4 x 16.78 MB bf16 = 67 MB
    unsigned short* kd = qd + elems;
    unsigned short* vd = kd + elems;
    unsigned short* yd = vd + elems;

    qkv_gemm_kernel<<<128 * 48, 128, 0, stream>>>(x, W_attn, b_attn, qd, kd, vd);
    attn_kernel<<<1024, 256, 0, stream>>>(qd, kd, vd, yd);
    proj_gemm_kernel<<<128 * 16, 128, 0, stream>>>(yd, W_proj, b_proj, out);
}
